// GATResidueEmbedding_65085934403599
// MI455X (gfx1250) — compile-verified, matrix operand order checked by reading, not run on hardware
//
#include <hip/hip_runtime.h>
#include <hip/hip_bf16.h>

typedef __attribute__((ext_vector_type(16))) _Float16 v16h;
typedef __attribute__((ext_vector_type(8)))  float    v8f;
typedef __attribute__((ext_vector_type(4)))  float    v4f;

// ---------- helpers ----------
static __device__ __forceinline__ unsigned orderF(float f) {
  unsigned u = __float_as_uint(f);
  return (u & 0x80000000u) ? ~u : (u | 0x80000000u);
}
static __device__ __forceinline__ float unorderF(unsigned u) {
  u = (u & 0x80000000u) ? (u & 0x7fffffffu) : ~u;
  return __uint_as_float(u);
}
static __device__ __forceinline__ void edge_sd(const int* __restrict__ ei, int E,
                                               long e, int& s, int& d) {
  if (e < (long)E) { s = ei[e]; d = ei[(long)E + e]; }
  else             { s = d = (int)(e - (long)E); }   // self loop
}

// ---------- 1. gather embedding rows -> f16 ----------
__global__ void gather_embed(const int* __restrict__ x, const float* __restrict__ emb,
                             _Float16* __restrict__ h0, int N) {
  int idx = blockIdx.x * blockDim.x + threadIdx.x;
  if (idx >= N * 128) return;
  int n = idx >> 7, c = idx & 127;
  h0[idx] = (_Float16)emb[(size_t)x[n] * 128 + c];
}

// ---------- 2. pack weight matrix [K,Ncols] f32 into WMMA B-fragment layout (f16) ----------
// Bp[((tn*KB + kb)*32 + lane)*16 + i] ; lane<16 holds col tn*16+lane, K-halves per ISA layout
__global__ void pack_w(const float* __restrict__ W, _Float16* __restrict__ Bp,
                       int K, int Ncols) {
  int KB = K >> 5;
  int total = (Ncols >> 4) * KB * 512;
  int idx = blockIdx.x * blockDim.x + threadIdx.x;
  if (idx >= total) return;
  int per_tile = KB * 512;
  int tn   = idx / per_tile;
  int rem  = idx % per_tile;
  int kb   = rem >> 9;
  int lane = (rem >> 4) & 31;
  int i    = rem & 15;
  int off  = (lane >= 16) ? 8 : 0;
  int k    = kb * 32 + (i < 8 ? i : i + 8) + off;
  int col  = tn * 16 + (lane & 15);
  Bp[idx]  = (_Float16)W[(size_t)k * Ncols + col];
}

// ---------- 3. WMMA GEMM: C[M,Ncols] = A[M,K](f16,row-major) * Bp(packed f16) ----------
__global__ void gemm_f16_wmma(const _Float16* __restrict__ A,
                              const _Float16* __restrict__ Bp,
                              float* __restrict__ C,
                              int M, int K, int Ncols) {
  int wave  = threadIdx.x >> 5;
  int lane  = threadIdx.x & 31;
  int wpb   = blockDim.x >> 5;
  int ntN   = Ncols >> 4;
  int ntM   = (M + 15) >> 4;
  int tile  = blockIdx.x * wpb + wave;
  if (tile >= ntM * ntN) return;                 // wave-uniform
  int tm = tile / ntN;
  int tn = tile % ntN;
  bool hi = lane >= 16;
  int row = tm * 16 + (lane & 15);
  int rc  = row < M - 1 ? row : M - 1;
  const _Float16* arow = A + (size_t)rc * K + (hi ? 8 : 0);
  const _Float16* bptr = Bp + ((size_t)tn * (K >> 5) * 32 + lane) * 16;
  v8f acc = {};
  for (int k0 = 0; k0 < K; k0 += 32) {
    v16h a, b;
    *(v4f*)&a       = *(const v4f*)(arow + k0);        // K halves 0..7 (+off)
    *((v4f*)&a + 1) = *(const v4f*)(arow + k0 + 16);   // K halves 16..23 (+off)
    b = *(const v16h*)bptr;
    bptr += 512;
    acc = __builtin_amdgcn_wmma_f32_16x16x32_f16(false, a, false, b,
                                                 (short)0, acc, false, false);
  }
  int col   = tn * 16 + (lane & 15);
  int rbase = tm * 16 + (hi ? 8 : 0);
  #pragma unroll
  for (int r = 0; r < 8; ++r) {
    int rr = rbase + r;
    if (rr < M) C[(size_t)rr * Ncols + col] = acc[r];
  }
}

// ---------- 4. per-node attention dots ----------
__global__ void node_att(const float* __restrict__ h, const float* __restrict__ att_src,
                         const float* __restrict__ att_dst, float* __restrict__ as_,
                         float* __restrict__ ad_, int N, int H) {
  int idx = blockIdx.x * blockDim.x + threadIdx.x;
  if (idx >= N * H) return;
  int n = idx / H, hd = idx % H;
  const float* row = h + (size_t)n * H * 64 + hd * 64;
  float s = 0.f, d = 0.f;
  #pragma unroll 4
  for (int c = 0; c < 64; ++c) {
    float v = row[c];
    s += v * att_src[hd * 64 + c];
    d += v * att_dst[hd * 64 + c];
  }
  as_[idx] = s; ad_[idx] = d;
}

// ---------- 5. edge passes ----------
__global__ void edge_alpha_max(const int* __restrict__ ei, int E, int N, int H,
                               const float* __restrict__ as_, const float* __restrict__ ad_,
                               float* __restrict__ alpha, unsigned* __restrict__ mmax) {
  long ET = (long)E + N;
  long idx = (long)blockIdx.x * blockDim.x + threadIdx.x;
  if (idx >= ET * H) return;
  long e = idx / H; int hd = (int)(idx % H);
  int s, d; edge_sd(ei, E, e, s, d);
  float v = as_[(size_t)s * H + hd] + ad_[(size_t)d * H + hd];
  v = v > 0.f ? v : 0.2f * v;                    // leaky_relu 0.2
  alpha[idx] = v;
  atomicMax(&mmax[(size_t)d * H + hd], orderF(v));
}

__global__ void edge_exp_sum(const int* __restrict__ ei, int E, int N, int H,
                             float* __restrict__ alpha, const unsigned* __restrict__ mmax,
                             float* __restrict__ den) {
  long ET = (long)E + N;
  long idx = (long)blockIdx.x * blockDim.x + threadIdx.x;
  if (idx >= ET * H) return;
  long e = idx / H; int hd = (int)(idx % H);
  int s, d; edge_sd(ei, E, e, s, d);
  float m = unorderF(mmax[(size_t)d * H + hd]);
  float ex = __expf(alpha[idx] - m);
  alpha[idx] = ex;
  atomicAdd(&den[(size_t)d * H + hd], ex);
}

__global__ void edge_norm(const int* __restrict__ ei, int E, int N, int H,
                          float* __restrict__ alpha, const float* __restrict__ den) {
  long ET = (long)E + N;
  long idx = (long)blockIdx.x * blockDim.x + threadIdx.x;
  if (idx >= ET * H) return;
  long e = idx / H; int hd = (int)(idx % H);
  int s, d; edge_sd(ei, E, e, s, d);
  alpha[idx] = alpha[idx] / (den[(size_t)d * H + hd] + 1e-16f);
}

// ---------- 6. weighted scatter-aggregate (roofline-dominant; L2-resident atomics) ----------
__global__ void edge_aggregate(const int* __restrict__ ei, int E, int N, int H,
                               const float* __restrict__ alpha, const float* __restrict__ h,
                               float* __restrict__ out) {
  int HC = H * 64;
  long total = ((long)E + N) * HC;
  long stride = (long)gridDim.x * blockDim.x;
  for (long idx = (long)blockIdx.x * blockDim.x + threadIdx.x; idx < total; idx += stride) {
    long e = idx / HC; int c = (int)(idx % HC); int hd = c >> 6;
    int s, d; edge_sd(ei, E, e, s, d);
    float w = alpha[e * H + hd];
    atomicAdd(&out[(size_t)d * HC + c], h[(size_t)s * HC + c] * w);
  }
}

// ---------- 7. bias + layernorm + leaky(0.01); optional f16 out / pooled atomics ----------
__global__ void post_ln(const float* __restrict__ in, const float* __restrict__ bias,
                        const float* __restrict__ g, const float* __restrict__ be,
                        _Float16* __restrict__ out16, float* __restrict__ pool,
                        const int* __restrict__ batch, int N, int HC) {
  int wave = (blockIdx.x * blockDim.x + threadIdx.x) >> 5;
  int lane = threadIdx.x & 31;
  if (wave >= N) return;                          // wave-uniform
  const float* row = in + (size_t)wave * HC;
  float vals[6];
  int k = 0; float s = 0.f;
  for (int c = lane; c < HC; c += 32) { float v = row[c] + bias[c]; vals[k++] = v; s += v; }
  for (int m = 16; m; m >>= 1) s += __shfl_xor(s, m, 32);
  float mu = s / (float)HC;
  k = 0; float vs = 0.f;
  for (int c = lane; c < HC; c += 32) { float dv = vals[k++] - mu; vs += dv * dv; }
  for (int m = 16; m; m >>= 1) vs += __shfl_xor(vs, m, 32);
  float rstd = rsqrtf(vs / (float)HC + 1e-5f);
  int gidx = (pool != nullptr) ? batch[wave] : 0;
  k = 0;
  for (int c = lane; c < HC; c += 32) {
    float v = (vals[k++] - mu) * rstd * g[c] + be[c];
    v = v > 0.f ? v : 0.01f * v;
    if (out16) out16[(size_t)wave * HC + c] = (_Float16)v;
    if (pool)  atomicAdd(&pool[gidx * HC + c], v);
  }
}

// ---------- 8. per-graph node counts ----------
__global__ void count_nodes(const int* __restrict__ batch, float* __restrict__ cnt, int N) {
  int n = blockIdx.x * blockDim.x + threadIdx.x;
  if (n >= N) return;
  atomicAdd(&cnt[batch[n]], 1.0f);
}

// ---------- 9. mean pool + final linear ----------
__global__ void final_lin(const float* __restrict__ pool, const float* __restrict__ cnt,
                          const float* __restrict__ linW, const float* __restrict__ linb,
                          float* __restrict__ out, int G, int HID) {
  int g = blockIdx.x * blockDim.x + threadIdx.x;
  if (g >= G) return;
  float c = cnt[g]; c = c > 1.f ? c : 1.f;
  float s = 0.f;
  for (int i = 0; i < HID; ++i) s += (pool[(size_t)g * HID + i] / c) * linW[i];
  out[g] = s + linb[0];
}

// ---------- launch ----------
extern "C" void kernel_launch(void* const* d_in, const int* in_sizes, int n_in,
                              void* d_out, int out_size, void* d_ws, size_t ws_size,
                              hipStream_t stream) {
  const int*   x     = (const int*)d_in[0];
  const int*   ei    = (const int*)d_in[1];
  const int*   batch = (const int*)d_in[2];
  const float* emb   = (const float*)d_in[3];
  const float* W1    = (const float*)d_in[4];
  const float* atts1 = (const float*)d_in[5];
  const float* attd1 = (const float*)d_in[6];
  const float* b1    = (const float*)d_in[7];
  const float* g1    = (const float*)d_in[8];
  const float* be1   = (const float*)d_in[9];
  const float* W2    = (const float*)d_in[10];
  const float* atts2 = (const float*)d_in[11];
  const float* attd2 = (const float*)d_in[12];
  const float* b2    = (const float*)d_in[13];
  const float* g2    = (const float*)d_in[14];
  const float* be2   = (const float*)d_in[15];
  const float* linW  = (const float*)d_in[16];
  const float* linb  = (const float*)d_in[17];
  float* out = (float*)d_out;

  const int N   = in_sizes[0];        // x is (N,1)
  const int E   = in_sizes[1] / 2;    // edge_index (2,E)
  const int G   = out_size;           // 64 graphs
  const int EMB = 128, HID = 64, H1 = 3;
  const int HC1 = H1 * HID;           // 192
  const long ET = (long)E + N;

  char* p = (char*)d_ws;
  auto alloc = [&](size_t bytes) -> void* {
    char* r = p; p += (bytes + 255) & ~(size_t)255; return (void*)r;
  };
  _Float16* h0   = (_Float16*)alloc((size_t)N * EMB * 2);
  _Float16* W1p  = (_Float16*)alloc((size_t)(HC1 / 16) * (EMB / 32) * 512 * 2);
  float*    h1   = (float*)   alloc((size_t)N * HC1 * 4);
  float*    as1  = (float*)   alloc((size_t)N * H1 * 4);
  float*    ad1  = (float*)   alloc((size_t)N * H1 * 4);
  unsigned* mm1  = (unsigned*)alloc((size_t)N * H1 * 4);
  float*    den1 = (float*)   alloc((size_t)N * H1 * 4);
  float*    al1  = (float*)   alloc((size_t)ET * H1 * 4);
  float*    out1 = (float*)   alloc((size_t)N * HC1 * 4);
  _Float16* h2   = (_Float16*)alloc((size_t)N * HC1 * 2);
  _Float16* W2p  = (_Float16*)alloc((size_t)(HID / 16) * (HC1 / 32) * 512 * 2);
  float*    h3   = (float*)   alloc((size_t)N * HID * 4);
  float*    as2  = (float*)   alloc((size_t)N * 4);
  float*    ad2  = (float*)   alloc((size_t)N * 4);
  unsigned* mm2  = (unsigned*)alloc((size_t)N * 4);
  float*    den2 = (float*)   alloc((size_t)N * 4);
  float*    al2  = (float*)   alloc((size_t)ET * 4);
  float*    out2 = (float*)   alloc((size_t)N * HID * 4);
  float*    pool = (float*)   alloc((size_t)G * HID * 4);
  float*    cnt  = (float*)   alloc((size_t)G * 4);
  (void)ws_size; (void)n_in;

  hipMemsetAsync(mm1,  0, (size_t)N * H1 * 4, stream);
  hipMemsetAsync(den1, 0, (size_t)N * H1 * 4, stream);
  hipMemsetAsync(out1, 0, (size_t)N * HC1 * 4, stream);
  hipMemsetAsync(mm2,  0, (size_t)N * 4, stream);
  hipMemsetAsync(den2, 0, (size_t)N * 4, stream);
  hipMemsetAsync(out2, 0, (size_t)N * HID * 4, stream);
  hipMemsetAsync(pool, 0, (size_t)G * HID * 4, stream);
  hipMemsetAsync(cnt,  0, (size_t)G * 4, stream);

  auto cdiv = [](long a, long b) -> unsigned { return (unsigned)((a + b - 1) / b); };
  const int B = 256;

  // ---- layer 1 ----
  gather_embed<<<cdiv((long)N * 128, B), B, 0, stream>>>(x, emb, h0, N);
  pack_w<<<cdiv((long)(HC1 / 16) * (EMB / 32) * 512, B), B, 0, stream>>>(W1, W1p, EMB, HC1);
  {
    long tiles = (long)((N + 15) / 16) * (HC1 / 16);
    gemm_f16_wmma<<<cdiv(tiles, 8), B, 0, stream>>>(h0, W1p, h1, N, EMB, HC1);
  }
  node_att<<<cdiv((long)N * H1, B), B, 0, stream>>>(h1, atts1, attd1, as1, ad1, N, H1);
  edge_alpha_max<<<cdiv(ET * H1, B), B, 0, stream>>>(ei, E, N, H1, as1, ad1, al1, mm1);
  edge_exp_sum<<<cdiv(ET * H1, B), B, 0, stream>>>(ei, E, N, H1, al1, mm1, den1);
  edge_norm<<<cdiv(ET * H1, B), B, 0, stream>>>(ei, E, N, H1, al1, den1);
  edge_aggregate<<<cdiv(ET * HC1, B), B, 0, stream>>>(ei, E, N, H1, al1, h1, out1);
  post_ln<<<cdiv((long)N * 32, B), B, 0, stream>>>(out1, b1, g1, be1, h2, nullptr, batch, N, HC1);

  // ---- layer 2 ----
  pack_w<<<cdiv((long)(HID / 16) * (HC1 / 32) * 512, B), B, 0, stream>>>(W2, W2p, HC1, HID);
  {
    long tiles = (long)((N + 15) / 16) * (HID / 16);
    gemm_f16_wmma<<<cdiv(tiles, 8), B, 0, stream>>>(h2, W2p, h3, N, HC1, HID);
  }
  node_att<<<cdiv((long)N, B), B, 0, stream>>>(h3, atts2, attd2, as2, ad2, N, 1);
  edge_alpha_max<<<cdiv(ET, B), B, 0, stream>>>(ei, E, N, 1, as2, ad2, al2, mm2);
  edge_exp_sum<<<cdiv(ET, B), B, 0, stream>>>(ei, E, N, 1, al2, mm2, den2);
  edge_norm<<<cdiv(ET, B), B, 0, stream>>>(ei, E, N, 1, al2, den2);
  edge_aggregate<<<cdiv(ET * HID, B), B, 0, stream>>>(ei, E, N, 1, al2, h3, out2);
  post_ln<<<cdiv((long)N * 32, B), B, 0, stream>>>(out2, b2, g2, be2, nullptr, pool, batch, N, HID);

  // ---- readout ----
  count_nodes<<<cdiv(N, B), B, 0, stream>>>(batch, cnt, N);
  final_lin<<<1, 64, 0, stream>>>(pool, cnt, linW, linb, out, G, HID);
}
